// Cross_attention_dl_91061896610498
// MI455X (gfx1250) — matrix operand order repaired and checked by reading, not compiled, not run on hardware
//
#include <hip/hip_runtime.h>
#include <hip/hip_bf16.h>

// ---------------------------------------------------------------------------
// MI455X (gfx1250): all matmuls via v_wmma_f32_16x16x32_bf16, all fragment
// traffic as 128-bit loads. V is produced transposed by its projection GEMM so
// the attention P.V B-fragment is contiguous.
// ---------------------------------------------------------------------------

typedef __attribute__((ext_vector_type(16))) __bf16 v16bf;
typedef __attribute__((ext_vector_type(8)))  __bf16 v8bf;
typedef __attribute__((ext_vector_type(8)))  float  v8f;

__device__ inline unsigned short f2bfbits(float f) {
    union { float f; unsigned int u; } x; x.f = f;
    unsigned int r = x.u + 0x7fffu + ((x.u >> 16) & 1u);   // round-to-nearest-even
    return (unsigned short)(r >> 16);
}

__device__ inline v16bf cat8(v8bf lo, v8bf hi) {
    return __builtin_shufflevector(lo, hi, 0, 1, 2, 3, 4, 5, 6, 7,
                                   8, 9, 10, 11, 12, 13, 14, 15);
}

__device__ inline v8f wmma_bf16(v16bf a, v16bf b, v8f c) {
    return __builtin_amdgcn_wmma_f32_16x16x32_bf16(
        false, a, false, b, (short)0, c, false, false);
}

// A fragment: 16x32 bf16 from row-major source (leading dim lda, elements).
// Lane L holds row M=L%16; lanes<16: K={0..7,16..23}; lanes>=16: +8.
// Requires 16-byte alignment of (base + row*lda + kh) — guaranteed by callers.
__device__ inline v16bf load_A_frag(const unsigned short* base, int lda, int lane) {
    int row = lane & 15;
    int kh  = (lane >> 4) << 3;                       // 0 or 8
    const v8bf* p = reinterpret_cast<const v8bf*>(base + (size_t)row * lda + kh);
    return cat8(p[0], p[2]);                          // K {kh..kh+7}, {16+kh..23+kh}
}

// B fragment: 32x16 where B[k][n] = src[n][k] (source row n is B column n).
// Lane holds column n=lane%16; lanes<16: K={0..7,16..23}; lanes>=16: +8.
__device__ inline v16bf load_B_frag(const unsigned short* base, int ldr, int lane) {
    int n  = lane & 15;
    int kb = (lane >> 4) << 3;                        // 0 or 8
    const v8bf* p = reinterpret_cast<const v8bf*>(base + (size_t)n * ldr + kb);
    return cat8(p[0], p[2]);
}

// ---------------------------------------------------------------------------
// Vectorized f32 -> bf16 convert (n must be a multiple of 8).
__global__ void convert_f32_to_bf16(const float* __restrict__ src,
                                    unsigned short* __restrict__ dst, size_t n) {
    size_t i = ((size_t)blockIdx.x * blockDim.x + threadIdx.x) * 8;
    size_t stride = (size_t)gridDim.x * blockDim.x * 8;
    for (; i < n; i += stride) {
        float4 a = *reinterpret_cast<const float4*>(src + i);
        float4 b = *reinterpret_cast<const float4*>(src + i + 4);
        unsigned short v[8] = { f2bfbits(a.x), f2bfbits(a.y), f2bfbits(a.z), f2bfbits(a.w),
                                f2bfbits(b.x), f2bfbits(b.y), f2bfbits(b.z), f2bfbits(b.w) };
        *reinterpret_cast<uint4*>(dst + i) = *reinterpret_cast<const uint4*>(v);
    }
}

// ---------------------------------------------------------------------------
// C[M,N] = A[M,K] x W[N,K]^T + bias[N]  (all bf16 operands, f32 accumulate).
// transOut==0: C row-major [M][N] (bf16).  transOut!=0: C transposed [N][M]
// (bf16) — per-lane contiguous 16-byte stores.
// Block: 256 threads = 8 waves arranged 2(M) x 4(N) -> 32x64 tile.
__global__ void __launch_bounds__(256)
gemm_bias_bf16(const unsigned short* __restrict__ A, const unsigned short* __restrict__ W,
               const float* __restrict__ bias, unsigned short* __restrict__ C,
               int M, int N, int K, int transOut) {
    int lane = threadIdx.x & 31;
    int wave = threadIdx.x >> 5;
    int row0 = blockIdx.y * 32 + (wave >> 2) * 16;
    int col0 = blockIdx.x * 64 + (wave & 3) * 16;

    v8f acc = {};
    for (int k0 = 0; k0 < K; k0 += 32) {
        v16bf a = load_A_frag(A + (size_t)row0 * K + k0, K, lane);
        v16bf b = load_B_frag(W + (size_t)col0 * K + k0, K, lane);
        acc = wmma_bf16(a, b, acc);
    }
    int n     = col0 + (lane & 15);
    float bn  = bias ? bias[n] : 0.0f;
    int rbase = row0 + ((lane >> 4) << 3);
    if (transOut) {
        unsigned short v[8];
#pragma unroll
        for (int i = 0; i < 8; ++i) v[i] = f2bfbits(acc[i] + bn);
        *reinterpret_cast<uint4*>(C + (size_t)n * M + rbase) =
            *reinterpret_cast<const uint4*>(v);
    } else {
#pragma unroll
        for (int i = 0; i < 8; ++i)
            C[(size_t)(rbase + i) * N + n] = f2bfbits(acc[i] + bn);
    }
}

// ---------------------------------------------------------------------------
// Flash attention, no 1/sqrt(d) scaling (faithful to reference).
// Q,K row-major per batch [S][D]; V transposed global [D][Mtotal] with this
// batch's columns at offset b*S. Grid: (S/16, batches); 256 threads = 8 waves.
template <int D>
__global__ void __launch_bounds__(256)
attn_kernel(const unsigned short* __restrict__ Qm, const unsigned short* __restrict__ Km,
            const unsigned short* __restrict__ Vt, int ldVt, int S,
            const float* __restrict__ resX, const float* __restrict__ resY,
            const float* __restrict__ w1, const float* __restrict__ w2,
            unsigned short* __restrict__ outBF, float* __restrict__ outF, int outLd) {
    constexpr int DW = D / 8;    // dims per wave
    constexpr int NT = DW / 16;  // 16-col accumulator tiles per wave
    constexpr int KB = 256;      // keys per block iteration

    __shared__ alignas(16) unsigned short sQ[16 * D];
    __shared__ alignas(16) float          sS[16 * KB];
    __shared__ alignas(16) unsigned short sP[16 * KB];
    __shared__ float sMax[16], sSum[16], sScale[16];

    int lane = threadIdx.x & 31;
    int wave = threadIdx.x >> 5;
    int b    = blockIdx.y;
    int q0   = blockIdx.x * 16;

    const unsigned short* Qb = Qm + ((size_t)b * S + q0) * D;
    const unsigned short* Kb = Km + (size_t)b * S * D;

    for (int idx = threadIdx.x * 8; idx < 16 * D; idx += 256 * 8)
        *reinterpret_cast<uint4*>(sQ + idx) = *reinterpret_cast<const uint4*>(Qb + idx);
    if (threadIdx.x < 16) { sMax[threadIdx.x] = -1e30f; sSum[threadIdx.x] = 0.0f; }
    __syncthreads();

    v8f acc[NT];
#pragma unroll
    for (int t = 0; t < NT; ++t)
#pragma unroll
        for (int i = 0; i < 8; ++i) acc[t][i] = 0.0f;

    for (int kb0 = 0; kb0 < S; kb0 += KB) {
        // ---- scores: this wave handles keys [kb0 + wave*32, +32) -----------
        int keyBase = kb0 + wave * 32;
        v8f s0 = {}, s1 = {};
        for (int k0 = 0; k0 < D; k0 += 32) {
            v16bf aq = load_A_frag(sQ + k0, D, lane);
            v16bf b0 = load_B_frag(Kb + (size_t)keyBase * D + k0, D, lane);
            v16bf b1 = load_B_frag(Kb + (size_t)(keyBase + 16) * D + k0, D, lane);
            s0 = wmma_bf16(aq, b0, s0);
            s1 = wmma_bf16(aq, b1, s1);
        }
        {
            int rb = (lane >> 4) << 3;
            int cn = lane & 15;
#pragma unroll
            for (int i = 0; i < 8; ++i) {
                sS[(rb + i) * KB + wave * 32 + cn]      = s0[i];
                sS[(rb + i) * KB + wave * 32 + 16 + cn] = s1[i];
            }
        }
        __syncthreads();

        // ---- online softmax: wave handles rows {2*wave, 2*wave+1} ----------
#pragma unroll
        for (int rr = 0; rr < 2; ++rr) {
            int r = wave * 2 + rr;
            const float* srow = sS + r * KB;
            float vals[8];
            float vmax = -1e30f;
#pragma unroll
            for (int j = 0; j < 8; ++j) { vals[j] = srow[lane + 32 * j]; vmax = fmaxf(vmax, vals[j]); }
#pragma unroll
            for (int off = 16; off > 0; off >>= 1) vmax = fmaxf(vmax, __shfl_xor(vmax, off, 32));
            float mprev = sMax[r];
            float mnew  = fmaxf(mprev, vmax);
            float psum  = 0.0f;
#pragma unroll
            for (int j = 0; j < 8; ++j) {
                float p = __expf(vals[j] - mnew);
                psum += p;
                sP[r * KB + lane + 32 * j] = f2bfbits(p);
            }
#pragma unroll
            for (int off = 16; off > 0; off >>= 1) psum += __shfl_xor(psum, off, 32);
            if (lane == 0) {
                float sc  = __expf(mprev - mnew);
                sScale[r] = sc;
                sSum[r]   = sSum[r] * sc + psum;
                sMax[r]   = mnew;
            }
        }
        __syncthreads();

        // ---- rescale accumulators, then P.V ------------------------------
        {
            int rb = (lane >> 4) << 3;
#pragma unroll
            for (int t = 0; t < NT; ++t)
#pragma unroll
                for (int i = 0; i < 8; ++i) acc[t][i] *= sScale[rb + i];
        }
        for (int sub = 0; sub < KB / 32; ++sub) {
            v16bf ap = load_A_frag(sP + sub * 32, KB, lane);
            size_t keyCol = (size_t)b * S + kb0 + sub * 32;
#pragma unroll
            for (int t = 0; t < NT; ++t) {
                // B[k][n] = Vt[dim0+n][keyCol+k] -> row n of Vt is B column n.
                v16bf bv = load_B_frag(
                    Vt + (size_t)(wave * DW + t * 16) * ldVt + keyCol, ldVt, lane);
                acc[t] = wmma_bf16(ap, bv, acc[t]);
            }
        }
        // Barriers above fence sS/sP reuse across iterations.
    }
    __syncthreads();

    // ---- epilogue ---------------------------------------------------------
    int  rb = (lane >> 4) << 3;
    int  cn = lane & 15;
    bool stage1 = (outBF != nullptr);
    const float* res = nullptr;
    float rscale = 0.0f;
    int outb = b, outCol = 0;
    if (stage1) {
        if (b < 4) { res = resY + ((size_t)b * S + q0) * D;       rscale = *w2; outb = b;     outCol = 0; }
        else       { res = resX + ((size_t)(b - 4) * S + q0) * D; rscale = *w1; outb = b - 4; outCol = D; }
    }
#pragma unroll
    for (int t = 0; t < NT; ++t) {
        int dim = wave * DW + t * 16 + cn;
#pragma unroll
        for (int i = 0; i < 8; ++i) {
            int   r = rb + i;
            float v = acc[t][i] / sSum[r];
            if (stage1) {
                v += res[(size_t)r * D + dim] * rscale;
                outBF[((size_t)outb * S + q0 + r) * outLd + outCol + dim] = f2bfbits(v);
            } else {
                outF[((size_t)b * S + q0 + r) * outLd + dim] = v;
            }
        }
    }
}

// ---------------------------------------------------------------------------
extern "C" void kernel_launch(void* const* d_in, const int* in_sizes, int n_in,
                              void* d_out, int out_size, void* d_ws, size_t ws_size,
                              hipStream_t stream) {
    (void)in_sizes; (void)n_in; (void)out_size; (void)ws_size;
    constexpr int D1 = 512, D2 = 1024, B = 4, S = 2048;

    const float* x   = (const float*)d_in[0];
    const float* y   = (const float*)d_in[1];
    const float* Wq1 = (const float*)d_in[2];
    const float* bq1 = (const float*)d_in[3];
    const float* Wk1 = (const float*)d_in[4];
    const float* bk1 = (const float*)d_in[5];
    const float* Wv1 = (const float*)d_in[6];
    const float* bv1 = (const float*)d_in[7];
    const float* Wq2 = (const float*)d_in[8];
    const float* bq2 = (const float*)d_in[9];
    const float* Wk2 = (const float*)d_in[10];
    const float* bk2 = (const float*)d_in[11];
    const float* Wv2 = (const float*)d_in[12];
    const float* bv2 = (const float*)d_in[13];
    const float* w1  = (const float*)d_in[14];
    const float* w2  = (const float*)d_in[15];
    float*       out = (float*)d_out;

    // Workspace (bf16 as ushort). Five 8M-element slabs (16 MB each) plus
    // pre-converted weights (~7.5 MB). Q/K/V slabs reused across stages
    // (8*2048*512 == 4*2048*1024).
    const size_t slab = (size_t)2 * B * S * D1;      // 8,388,608 elements
    unsigned short* XY   = (unsigned short*)d_ws;
    unsigned short* QA   = XY + slab;
    unsigned short* KA   = QA + slab;
    unsigned short* VA   = KA + slab;                 // holds V transposed
    unsigned short* TEMP = VA + slab;
    unsigned short* Wq1b = TEMP + slab;
    unsigned short* Wk1b = Wq1b + (size_t)D1 * D1;
    unsigned short* Wv1b = Wk1b + (size_t)D1 * D1;
    unsigned short* Wq2b = Wv1b + (size_t)D1 * D1;
    unsigned short* Wk2b = Wq2b + (size_t)D2 * D2;
    unsigned short* Wv2b = Wk2b + (size_t)D2 * D2;

    const size_t half = (size_t)B * S * D1;           // 4M elements each

    // 1) convert activations and all weights to bf16
    convert_f32_to_bf16<<<1024, 256, 0, stream>>>(x, XY, half);
    convert_f32_to_bf16<<<1024, 256, 0, stream>>>(y, XY + half, half);
    convert_f32_to_bf16<<<256, 256, 0, stream>>>(Wq1, Wq1b, (size_t)D1 * D1);
    convert_f32_to_bf16<<<256, 256, 0, stream>>>(Wk1, Wk1b, (size_t)D1 * D1);
    convert_f32_to_bf16<<<256, 256, 0, stream>>>(Wv1, Wv1b, (size_t)D1 * D1);
    convert_f32_to_bf16<<<512, 256, 0, stream>>>(Wq2, Wq2b, (size_t)D2 * D2);
    convert_f32_to_bf16<<<512, 256, 0, stream>>>(Wk2, Wk2b, (size_t)D2 * D2);
    convert_f32_to_bf16<<<512, 256, 0, stream>>>(Wv2, Wv2b, (size_t)D2 * D2);

    // 2) stage-1 QKV projections: M=16384, N=K=512 (V written transposed)
    {
        const int M = 2 * B * S;
        dim3 grid(D1 / 64, M / 32);
        gemm_bias_bf16<<<grid, 256, 0, stream>>>(XY, Wq1b, bq1, QA, M, D1, D1, 0);
        gemm_bias_bf16<<<grid, 256, 0, stream>>>(XY, Wk1b, bk1, KA, M, D1, D1, 0);
        gemm_bias_bf16<<<grid, 256, 0, stream>>>(XY, Wv1b, bv1, VA, M, D1, D1, 1);
    }

    // 3) stage-1 attention: b<4 -> attn(x)+y*w2 into temp[:, :512],
    //                       b>=4 -> attn(y)+x*w1 into temp[:, 512:]
    attn_kernel<D1><<<dim3(S / 16, 2 * B), 256, 0, stream>>>(
        QA, KA, VA, 2 * B * S, S, x, y, w1, w2, TEMP, nullptr, D2);

    // 4) stage-2 QKV projections on temp: M=8192, N=K=1024 (reuse slabs)
    {
        const int M = B * S;
        dim3 grid(D2 / 64, M / 32);
        gemm_bias_bf16<<<grid, 256, 0, stream>>>(TEMP, Wq2b, bq2, QA, M, D2, D2, 0);
        gemm_bias_bf16<<<grid, 256, 0, stream>>>(TEMP, Wk2b, bk2, KA, M, D2, D2, 0);
        gemm_bias_bf16<<<grid, 256, 0, stream>>>(TEMP, Wv2b, bv2, VA, M, D2, D2, 1);
    }

    // 5) stage-2 attention -> f32 output [4][2048][1024]
    attn_kernel<D2><<<dim3(S / 16, B), 256, 0, stream>>>(
        QA, KA, VA, B * S, S, nullptr, nullptr, nullptr, nullptr, nullptr, out, D2);
}
